// QuantLlamaAttention_8117488189486
// MI455X (gfx1250) — compile-verified, matrix operand order checked by reading, not run on hardware
//
#include <hip/hip_runtime.h>
#include <cstdint>
#include <cstddef>

#define B   2
#define S   1024
#define HID 4096
#define NH  32
#define DH  128
#define MROWS (B * S)
#define NEG_INF (-3.4028234663852886e38f)

typedef __attribute__((ext_vector_type(8)))  int       v8i;
typedef __attribute__((ext_vector_type(8)))  float     v8f;
typedef __attribute__((ext_vector_type(8)))  _Float16  v8h;
typedef __attribute__((ext_vector_type(16))) _Float16  v16h;

__device__ __forceinline__ void async_copy_b128(const int8_t* g, int8_t* l) {
  const unsigned loff =
      (unsigned)(size_t)(const __attribute__((address_space(3))) int8_t*)l;
  asm volatile("global_load_async_to_lds_b128 %0, %1, off"
               :: "v"(loff), "v"(g) : "memory");
}
__device__ __forceinline__ void wait_asynccnt0() {
  asm volatile("s_wait_asynccnt 0" ::: "memory");
}

__global__ __launch_bounds__(256)
void k_quant_rows(const float* __restrict__ X, int8_t* __restrict__ Q8,
                  float* __restrict__ Sout, int cols, int qbits) {
  __shared__ float red[256];
  const int row = blockIdx.x;
  const float qmax = (float)((1 << (qbits - 1)) - 1);
  const float qmin = -qmax - 1.0f;
  const float* xr = X + (size_t)row * cols;
  float mx = 0.0f;
  for (int c = threadIdx.x; c < cols; c += 256)
    mx = fmaxf(mx, fabsf(xr[c]));
  red[threadIdx.x] = mx; __syncthreads();
  for (int s = 128; s > 0; s >>= 1) {
    if (threadIdx.x < s) red[threadIdx.x] = fmaxf(red[threadIdx.x], red[threadIdx.x + s]);
    __syncthreads();
  }
  const float scale = fmaxf(red[0] / qmax, 1e-8f);
  if (threadIdx.x == 0) Sout[row] = scale;
  int8_t* qr = Q8 + (size_t)row * cols;
  const float inv = 1.0f / scale;
  for (int c = threadIdx.x; c < cols; c += 256) {
    float n = rintf(xr[c] * inv);
    n = fminf(fmaxf(n, qmin), qmax);
    qr[c] = (int8_t)(int)n;
  }
}

__global__ __launch_bounds__(256)
void k_gemm_iu8(const int8_t* __restrict__ A, const int8_t* __restrict__ Bw,
                const float* __restrict__ sa, const float* __restrict__ sb,
                float* __restrict__ C, int M, int N, int K) {
  __shared__ int8_t As[128 * 256];
  __shared__ int8_t Bs[128 * 256];
  const int m0   = blockIdx.y * 128;
  const int n0   = blockIdx.x * 128;
  const int wave = (int)threadIdx.x / 32;
  const int wm   = wave >> 2;
  const int wn   = wave & 3;
  const int lane = threadIdx.x & 31;
  const int half = lane >> 4;
  const int l16  = lane & 15;

  v8i acc[4][2];
#pragma unroll
  for (int mt = 0; mt < 4; ++mt)
#pragma unroll
    for (int nt = 0; nt < 2; ++nt)
      acc[mt][nt] = (v8i){0, 0, 0, 0, 0, 0, 0, 0};

  const int ldrow = (int)threadIdx.x >> 1;
  const int ldseg = ((int)threadIdx.x & 1) * 128;

  for (int kc = 0; kc < K; kc += 256) {
#pragma unroll
    for (int i = 0; i < 8; ++i) {
      async_copy_b128(A + (size_t)(m0 + ldrow) * K + kc + ldseg + i * 16,
                      As + ldrow * 256 + ldseg + i * 16);
      async_copy_b128(Bw + (size_t)(n0 + ldrow) * K + kc + ldseg + i * 16,
                      Bs + ldrow * 256 + ldseg + i * 16);
    }
    if (kc + 256 < K) {
      __builtin_prefetch(A + (size_t)(m0 + ldrow) * K + kc + 256 + ldseg, 0, 0);
      __builtin_prefetch(Bw + (size_t)(n0 + ldrow) * K + kc + 256 + ldseg, 0, 0);
    }
    wait_asynccnt0();
    __syncthreads();
#pragma unroll
    for (int ks = 0; ks < 4; ++ks) {
      v8i af[4], bf[2];
#pragma unroll
      for (int mt = 0; mt < 4; ++mt)
        af[mt] = *(const v8i*)(As + (wm * 64 + mt * 16 + l16) * 256 + ks * 64 + half * 32);
#pragma unroll
      for (int nt = 0; nt < 2; ++nt)
        bf[nt] = *(const v8i*)(Bs + (wn * 32 + nt * 16 + l16) * 256 + ks * 64 + half * 32);
#pragma unroll
      for (int mt = 0; mt < 4; ++mt)
#pragma unroll
        for (int nt = 0; nt < 2; ++nt)
          acc[mt][nt] = __builtin_amdgcn_wmma_i32_16x16x64_iu8(
              true, af[mt], true, bf[nt], acc[mt][nt], false, false);
    }
    __syncthreads();
  }

#pragma unroll
  for (int mt = 0; mt < 4; ++mt)
#pragma unroll
    for (int nt = 0; nt < 2; ++nt)
#pragma unroll
      for (int r = 0; r < 8; ++r) {
        const int m = m0 + wm * 64 + mt * 16 + r + 8 * half;
        const int n = n0 + wn * 32 + nt * 16 + l16;
        C[(size_t)m * N + n] = (float)acc[mt][nt][r] * sa[m] * sb[n];
      }
}

__global__ __launch_bounds__(128)
void k_rope_quant(const float* __restrict__ X, int8_t* __restrict__ Q8,
                  _Float16* __restrict__ VT, float* __restrict__ coef, int mode) {
  __shared__ float red[128];
  const int bs = blockIdx.x;
  const int s  = bs % S;
  const int h  = blockIdx.y;
  const int d  = threadIdx.x;
  const size_t base = (size_t)bs * HID + (size_t)h * DH;

  float val = X[base + d];
  if (mode < 2) {
    const int j = d & 63;
    const float f = (float)s * __expf(-9.210340371976184f * ((float)(2 * j) / 128.0f));
    const float rot = (d < 64) ? -X[base + d + 64] : X[base + d - 64];
    val = val * __cosf(f) + rot * __sinf(f);
  }

  red[d] = fabsf(val); __syncthreads();
  for (int t = 64; t > 0; t >>= 1) {
    if (d < t) red[d] = fmaxf(red[d], red[d + t]);
    __syncthreads();
  }
  const float scale = fmaxf(red[0] / 127.0f, 1e-8f);
  __syncthreads();

  red[d] = val * val; __syncthreads();
  for (int t = 64; t > 0; t >>= 1) {
    if (d < t) red[d] += red[d + t];
    __syncthreads();
  }
  const float normx = sqrtf(red[0]) + 1e-20f;
  __syncthreads();

  float n = rintf(val / scale);
  n = fminf(fmaxf(n, -128.0f), 127.0f);

  red[d] = n * n; __syncthreads();
  for (int t = 64; t > 0; t >>= 1) {
    if (d < t) red[d] += red[d + t];
    __syncthreads();
  }
  const float normn = sqrtf(red[0]) + 1e-20f;

  const int bh = (bs / S) * NH + h;
  if (mode == 2) {
    VT[((size_t)bh * DH + d) * S + s] = (_Float16)(n * scale);
    if (d == 0) coef[(size_t)bh * S + s] = (scale * normn) / normx;
  } else {
    Q8[base + d] = (int8_t)(int)n;
    if (d == 0) {
      float c = normx / normn;
      if (mode == 0) c *= 0.08838834764831845f;
      coef[(size_t)bh * S + s] = c;
    }
  }
}

__global__ __launch_bounds__(32)
void k_attn(const int8_t* __restrict__ Q8, const int8_t* __restrict__ K8,
            const _Float16* __restrict__ VT, const float* __restrict__ cq,
            const float* __restrict__ ck, const float* __restrict__ a2v,
            float* __restrict__ O) {
  extern __shared__ float scores[];
  const int bh   = blockIdx.x;
  const int b    = bh / NH;
  const int h    = bh % NH;
  const int qt   = blockIdx.y;
  const int q0   = qt * 16;
  const int lane = threadIdx.x;
  const int half = lane >> 4;
  const int l16  = lane & 15;

  const int8_t* qp = Q8 + ((size_t)(b * S + q0 + l16) * NH + h) * DH + half * 32;
  const v8i qa0 = *(const v8i*)qp;
  const v8i qa1 = *(const v8i*)(qp + 64);

  float cqr[8];
#pragma unroll
  for (int r = 0; r < 8; ++r) cqr[r] = cq[(size_t)bh * S + q0 + r + 8 * half];

  for (int j = 0; j <= qt; ++j) {
    const int kbase = j * 16;
    const int8_t* kp = K8 + ((size_t)(b * S + kbase + l16) * NH + h) * DH + half * 32;
    const v8i kb0 = *(const v8i*)kp;
    const v8i kb1 = *(const v8i*)(kp + 64);
    v8i si = {0, 0, 0, 0, 0, 0, 0, 0};
    si = __builtin_amdgcn_wmma_i32_16x16x64_iu8(true, qa0, true, kb0, si, false, false);
    si = __builtin_amdgcn_wmma_i32_16x16x64_iu8(true, qa1, true, kb1, si, false, false);
    const float ckc = ck[(size_t)bh * S + kbase + l16];
#pragma unroll
    for (int r = 0; r < 8; ++r) {
      const int row = r + 8 * half;
      const int kg  = kbase + l16;
      scores[row * S + kg] =
          (kg <= q0 + row) ? (float)si[r] * cqr[r] * ckc : NEG_INF;
    }
  }
  __syncthreads();

  const int kmax = (qt + 1) * 16;
  const int kp32 = (kmax + 31) & ~31;
  if (lane < 16) {
    float* sr = scores + lane * S;
    float m = NEG_INF;
    for (int k = 0; k < kmax; ++k) m = fmaxf(m, sr[k]);
    float sum = 0.0f;
    for (int k = 0; k < kmax; ++k) { const float e = __expf(sr[k] - m); sr[k] = e; sum += e; }
    const float invsum = 1.0f / sum;
    const float sp = fmaxf(invsum / 127.0f, 1e-8f);
    for (int k = 0; k < kmax; ++k) {
      float nq = rintf(sr[k] * invsum / sp);
      nq = fminf(nq, 127.0f);
      sr[k] = nq * sp;
    }
    for (int k = kmax; k < kp32; ++k) sr[k] = 0.0f;
  }
  __syncthreads();

  v8f o[8];
#pragma unroll
  for (int t = 0; t < 8; ++t) o[t] = (v8f){0, 0, 0, 0, 0, 0, 0, 0};

  for (int kb = 0; kb < kp32 / 32; ++kb) {
    const int base = kb * 32;
    v16h pa;
    const float* pr = scores + l16 * S + base;
#pragma unroll
    for (int jj = 0; jj < 8; ++jj) {
      pa[jj]     = (_Float16)pr[half * 8 + jj];
      pa[8 + jj] = (_Float16)pr[16 + half * 8 + jj];
    }
#pragma unroll
    for (int t = 0; t < 8; ++t) {
      const _Float16* vp = VT + ((size_t)bh * DH + t * 16 + l16) * S + base;
      const v8h lo = *(const v8h*)(vp + half * 8);
      const v8h hi = *(const v8h*)(vp + 16 + half * 8);
      const v16h vb = __builtin_shufflevector(lo, hi, 0, 1, 2, 3, 4, 5, 6, 7,
                                              8, 9, 10, 11, 12, 13, 14, 15);
      o[t] = __builtin_amdgcn_wmma_f32_16x16x32_f16(false, pa, false, vb,
                                                    (short)0, o[t], false, false);
    }
  }

#pragma unroll
  for (int r = 0; r < 8; ++r) {
    const int row = r + 8 * half;
    const float inv = 1.0f / a2v[(size_t)bh * S + q0 + row];
#pragma unroll
    for (int t = 0; t < 8; ++t)
      O[((size_t)(b * S + q0 + row) * NH + h) * DH + t * 16 + l16] = o[t][r] * inv;
  }
}

extern "C" void kernel_launch(void* const* d_in, const int* in_sizes, int n_in,
                              void* d_out, int out_size, void* d_ws, size_t ws_size,
                              hipStream_t stream) {
  (void)in_sizes; (void)n_in; (void)out_size; (void)ws_size;
  const float* hidden = (const float*)d_in[0];
  const float* w[4]   = {(const float*)d_in[1], (const float*)d_in[2],
                         (const float*)d_in[3], (const float*)d_in[4]};
  float* out = (float*)d_out;

  char* p = (char*)d_ws;
  auto carve = [&](size_t bytes) -> void* {
    void* r = (void*)p;
    p += (bytes + 255) & ~(size_t)255;
    return r;
  };
  int8_t* qx   = (int8_t*)carve((size_t)MROWS * HID);
  float*  sx   = (float*)carve((size_t)MROWS * 4);
  int8_t* qw   = (int8_t*)carve((size_t)4 * HID * HID);
  float*  sw   = (float*)carve((size_t)4 * HID * 4);
  float*  projQ = (float*)carve((size_t)MROWS * HID * 4);
  float*  projK = (float*)carve((size_t)MROWS * HID * 4);
  float*  projV = (float*)carve((size_t)MROWS * HID * 4);
  int8_t* q8   = (int8_t*)carve((size_t)MROWS * HID);
  int8_t* k8   = (int8_t*)carve((size_t)MROWS * HID);
  float*  cq   = (float*)carve((size_t)B * NH * S * 4);
  float*  ck   = (float*)carve((size_t)B * NH * S * 4);
  _Float16* vT = (_Float16*)carve((size_t)B * NH * DH * S * 2);
  float*  a2v  = (float*)carve((size_t)B * NH * S * 4);
  float*  attn = (float*)carve((size_t)MROWS * HID * 4);
  int8_t* qo   = (int8_t*)carve((size_t)MROWS * HID);
  float*  so   = (float*)carve((size_t)MROWS * 4);

  k_quant_rows<<<MROWS, 256, 0, stream>>>(hidden, qx, sx, HID, 8);
  for (int i = 0; i < 4; ++i)
    k_quant_rows<<<HID, 256, 0, stream>>>(w[i], qw + (size_t)i * HID * HID,
                                          sw + (size_t)i * HID, HID, 4);

  const dim3 ggrid(HID / 128, MROWS / 128);
  k_gemm_iu8<<<ggrid, 256, 0, stream>>>(qx, qw + 0 * (size_t)HID * HID, sx, sw + 0 * HID,
                                        projQ, MROWS, HID, HID);
  k_gemm_iu8<<<ggrid, 256, 0, stream>>>(qx, qw + 1 * (size_t)HID * HID, sx, sw + 1 * HID,
                                        projK, MROWS, HID, HID);
  k_gemm_iu8<<<ggrid, 256, 0, stream>>>(qx, qw + 2 * (size_t)HID * HID, sx, sw + 2 * HID,
                                        projV, MROWS, HID, HID);

  const dim3 rgrid(MROWS, NH);
  k_rope_quant<<<rgrid, 128, 0, stream>>>(projQ, q8, nullptr, cq, 0);
  k_rope_quant<<<rgrid, 128, 0, stream>>>(projK, k8, nullptr, ck, 1);
  k_rope_quant<<<rgrid, 128, 0, stream>>>(projV, nullptr, vT, a2v, 2);

  k_attn<<<dim3(B * NH, S / 16), 32, 16 * S * sizeof(float), stream>>>(
      q8, k8, vT, cq, ck, a2v, attn);

  k_quant_rows<<<MROWS, 256, 0, stream>>>(attn, qo, so, HID, 8);
  k_gemm_iu8<<<ggrid, 256, 0, stream>>>(qo, qw + 3 * (size_t)HID * HID, so, sw + 3 * HID,
                                        out, MROWS, HID, HID);
}
